// SyncNllbMoeSparseMLP_77266461655593
// MI455X (gfx1250) — compile-verified, matrix operand order checked by reading, not run on hardware
//
#include <hip/hip_runtime.h>
#include <hip/hip_bf16.h>
#include <math.h>

// Problem constants (reference: B=2, S=1024, D=1024, F=2048, E=8)
constexpr int NT = 2048;   // tokens = B*S
constexpr int ND = 1024;   // model dim
constexpr int NF = 2048;   // ffn dim
constexpr int NE = 8;      // experts
constexpr int TM = 32;     // tokens per workgroup tile
constexpr int TILES = NT / TM;   // tiles per expert (capacity = NT)

// dynamic LDS layout (bytes)
constexpr int XS_BYTES  = TM * ND * 2;          // 65536 : x tile f16
constexpr int HS_BYTES  = TM * 128 * 2;         // 8192  : h chunk f16
constexpr int WS1_BYTES = 2 * 128 * 16 * 4;     // 16384 : w1 panel dbl-buf (uint)
constexpr int META_BYTES = TM * 4 * 2;          // stok + sgate
constexpr int SMEM_BYTES = XS_BYTES + HS_BYTES + WS1_BYTES + META_BYTES; // ~90.4 KB

typedef __attribute__((ext_vector_type(16))) _Float16 v16h;
typedef __attribute__((ext_vector_type(8)))  float    v8f;

union frag16 { v16h h; unsigned u[8]; };

__device__ __forceinline__ unsigned pack2h(float a, float b) {
    union { _Float16 h[2]; unsigned u; } t;
    t.h[0] = (_Float16)a;   // low 16 bits -> even K element
    t.h[1] = (_Float16)b;   // high 16 bits -> odd K element
    return t.u;
}

// K value of the EVEN element of pair v (0..7) in a 16x16x32 f16 fragment,
// for lanes in wave-half hlf. (lanes 0-15: K=0..7,16..23; lanes 16-31: +8)
__device__ __forceinline__ int kpair_base(int v, int hlf) {
    return (v < 4) ? (2 * v + 8 * hlf) : (16 + 2 * (v - 4) + 8 * hlf);
}

__device__ __forceinline__ v8f wmma_f16(v16h a, v16h b, v8f c) {
    return __builtin_amdgcn_wmma_f32_16x16x32_f16(
        false, a, false, b, (short)0, c, false, false);
}

// ---------------------------------------------------------------------------
// Kernel 1: router. One wave32 per token.
// ---------------------------------------------------------------------------
__global__ __launch_bounds__(256) void moe_router_kernel(
    const float* __restrict__ x, const float* __restrict__ rw,
    float* __restrict__ gatesOut, float* __restrict__ top1Out,
    int* __restrict__ cnt, int* __restrict__ pairTok, float* __restrict__ pairGate)
{
    const int lane = threadIdx.x & 31;
    const int wv   = threadIdx.x >> 5;
    const int t = blockIdx.x * 8 + wv;
    if (t >= NT) return;

    float acc[NE];
#pragma unroll
    for (int e = 0; e < NE; ++e) acc[e] = 0.f;

    const float* xr = x + (size_t)t * ND;
    for (int d = lane; d < ND; d += 32) {
        float xv = xr[d];
        const float* wr = rw + (size_t)d * NE;
#pragma unroll
        for (int e = 0; e < NE; ++e) acc[e] += xv * wr[e];
    }
#pragma unroll
    for (int off = 16; off >= 1; off >>= 1) {
#pragma unroll
        for (int e = 0; e < NE; ++e) acc[e] += __shfl_xor(acc[e], off, 32);
    }

    if (lane == 0) {
        float mx = acc[0];
#pragma unroll
        for (int e = 1; e < NE; ++e) mx = fmaxf(mx, acc[e]);
        float p[NE]; float s = 0.f;
#pragma unroll
        for (int e = 0; e < NE; ++e) { p[e] = __expf(acc[e] - mx); s += p[e]; }
        float inv = 1.f / s;
#pragma unroll
        for (int e = 0; e < NE; ++e) p[e] *= inv;

        int i1 = 0; float bp = p[0];
#pragma unroll
        for (int e = 1; e < NE; ++e) if (p[e] > bp) { bp = p[e]; i1 = e; }
        int i2 = (i1 == 0) ? 1 : 0; float bl = acc[i2];
#pragma unroll
        for (int e = 0; e < NE; ++e)
            if (e != i1 && acc[e] > bl) { bl = acc[e]; i2 = e; }

        float p1 = p[i1], p2 = p[i2];
        float den = fmaxf(p1 + p2, 1.1920929e-7f);
        float g1 = p1 / den, g2 = p2 / den;

        float* go = gatesOut + (size_t)t * NE;
#pragma unroll
        for (int e = 0; e < NE; ++e)
            go[e] = (e == i1) ? g1 : ((e == i2) ? g2 : 0.f);
        top1Out[t] = (float)i1;

        int pos1 = atomicAdd(&cnt[i1], 1);
        pairTok[i1 * NT + pos1] = t; pairGate[i1 * NT + pos1] = g1;
        int pos2 = atomicAdd(&cnt[i2], 1);
        pairTok[i2 * NT + pos2] = t; pairGate[i2 * NT + pos2] = g2;
    }
}

// ---------------------------------------------------------------------------
// Kernel 2: fused per-expert FFN on gathered 32-token tiles.
// 8 waves / WG; wave w owns output columns [w*128, w*128+128) of D, for both
// 16-row M-subtiles. w1 K-panels double-buffered through LDS; every WMMA
// fragment read except w2 B (loaded from global) is b32/b128 from LDS; each w2 B fragment feeds 2 WMMAs.
// ---------------------------------------------------------------------------
__global__ __launch_bounds__(256) void moe_ffn_kernel(
    const float* __restrict__ x,
    const float* __restrict__ w1, const float* __restrict__ b1,
    const float* __restrict__ w2, const float* __restrict__ b2,
    const int* __restrict__ cnt, const int* __restrict__ pairTok,
    const float* __restrict__ pairGate, float* __restrict__ out)
{
    extern __shared__ char smem[];
    _Float16* xs   = (_Float16*)smem;                          // [TM][ND]
    _Float16* hs   = (_Float16*)(smem + XS_BYTES);             // [TM][128]
    unsigned* ws1  = (unsigned*)(smem + XS_BYTES + HS_BYTES);  // [2][128][16]
    int*      stok = (int*)(smem + XS_BYTES + HS_BYTES + WS1_BYTES);
    float*    sgate= (float*)(smem + XS_BYTES + HS_BYTES + WS1_BYTES + TM * 4);

    const int e    = blockIdx.x / TILES;
    const int tile = blockIdx.x % TILES;
    const int count = cnt[e];
    if (tile * TM >= count) return;

    const int tid  = threadIdx.x;
    const int lane = tid & 31;
    const int wv   = tid >> 5;       // 0..7
    const int hlf  = lane >> 4;      // wave half
    const int ln   = lane & 15;      // row/col within 16-tile

    if (tid < TM) {
        int pos = tile * TM + tid;
        if (pos < count) { stok[tid] = pairTok[e * NT + pos]; sgate[tid] = pairGate[e * NT + pos]; }
        else             { stok[tid] = -1;                    sgate[tid] = 0.f; }
    }
    __syncthreads();

    // Stage x tile (gathered rows) into LDS as f16 with b128 global loads.
    {
        const float4* x4 = (const float4*)x;
        for (int idx = tid; idx < TM * (ND / 4); idx += 256) {
            int r  = idx >> 8;          // / (ND/4) == /256
            int c4 = idx & 255;
            float4 v = {};
            int tok = stok[r];
            if (tok >= 0) v = x4[(size_t)tok * (ND / 4) + c4];
            unsigned* dst = (unsigned*)&xs[r * ND + c4 * 4];
            dst[0] = pack2h(v.x, v.y);
            dst[1] = pack2h(v.z, v.w);
        }
    }
    __syncthreads();

    const float* w1e = w1 + (size_t)e * ND * NF;
    const float* w2e = w2 + (size_t)e * NF * ND;

    // Cooperative stage of a 32K x 128F w1 panel into ws1[buf] (K-pair transposed).
    auto stageW1 = [&](int buf, int kbase, int f0) {
#pragma unroll
        for (int j = 0; j < 2; ++j) {
            int li = tid + j * 256;      // 0..511
            int k2 = li >> 5;            // K-pair index 0..15
            int f4 = li & 31;            // float4 index within F row
            const float* r0 = &w1e[(size_t)(kbase + 2 * k2) * NF + f0 + f4 * 4];
            float4 a = *(const float4*)r0;
            float4 b = *(const float4*)(r0 + NF);
            unsigned* dst = &ws1[((size_t)buf * 128) * 16];
            dst[(f4 * 4 + 0) * 16 + k2] = pack2h(a.x, b.x);
            dst[(f4 * 4 + 1) * 16 + k2] = pack2h(a.y, b.y);
            dst[(f4 * 4 + 2) * 16 + k2] = pack2h(a.z, b.z);
            dst[(f4 * 4 + 3) * 16 + k2] = pack2h(a.w, b.w);
        }
    };

    v8f zero = {};
    v8f acc[2][8];
#pragma unroll
    for (int ms = 0; ms < 2; ++ms)
#pragma unroll
        for (int nt = 0; nt < 8; ++nt) acc[ms][nt] = zero;

    const int nbase = wv * 128;
    const int fl    = wv * 16 + ln;   // this wave's local f column (0..127)

    for (int f0 = 0; f0 < NF; f0 += 128) {
        // prefetch next chunk's w2 rows (global_prefetch_b8)
        if (f0 + 128 < NF)
            __builtin_prefetch(&w2e[(size_t)(f0 + 128) * ND + nbase + ln], 0, 1);

        // -------- GEMM1: h[:, f0+fl] over K = D, w1 panel via LDS ---------
        stageW1(0, 0, f0);
        v8f hacc[2] = { zero, zero };
        for (int ks = 0; ks < ND / 32; ++ks) {
            const int buf = ks & 1;
            __syncthreads();                       // panel[buf] ready
            if (ks + 1 < ND / 32) stageW1(buf ^ 1, (ks + 1) * 32, f0);

            frag16 B;
            const int k0u = ks * 16;               // uint index base (32 halfs)
#pragma unroll
            for (int v = 0; v < 8; ++v)
                B.u[v] = ws1[(buf * 128 + fl) * 16 + (kpair_base(v, hlf) >> 1)];
#pragma unroll
            for (int ms = 0; ms < 2; ++ms) {
                frag16 A;
                const unsigned* xrow = (const unsigned*)&xs[(ms * 16 + ln) * ND];
#pragma unroll
                for (int v = 0; v < 8; ++v)
                    A.u[v] = xrow[k0u + (kpair_base(v, hlf) >> 1)];
                hacc[ms] = wmma_f16(A.h, B.h, hacc[ms]);
            }
        }

        // bias + relu -> hs
        float bias = b1[e * NF + f0 + fl];
#pragma unroll
        for (int ms = 0; ms < 2; ++ms)
#pragma unroll
            for (int r = 0; r < 8; ++r) {
                int m = ms * 16 + r + 8 * hlf;
                hs[m * 128 + fl] = (_Float16)fmaxf(hacc[ms][r] + bias, 0.f);
            }
        __syncthreads();

        // -------- GEMM2: accumulate y slices over this F chunk ------------
#pragma unroll
        for (int kk = 0; kk < 128; kk += 32) {
            frag16 A2[2];
#pragma unroll
            for (int ms = 0; ms < 2; ++ms) {
                const unsigned* hrow = (const unsigned*)&hs[(ms * 16 + ln) * 128];
#pragma unroll
                for (int v = 0; v < 8; ++v)
                    A2[ms].u[v] = hrow[(kk >> 1) + (kpair_base(v, hlf) >> 1)];
            }
#pragma unroll
            for (int nt = 0; nt < 8; ++nt) {
                frag16 B2;
                const int ncol = nbase + nt * 16 + ln;
#pragma unroll
                for (int v = 0; v < 8; ++v) {
                    int kg = f0 + kk + kpair_base(v, hlf);
                    const float* p = &w2e[(size_t)kg * ND + ncol];
                    B2.u[v] = pack2h(p[0], p[ND]);
                }
                acc[0][nt] = wmma_f16(A2[0].h, B2.h, acc[0][nt]);
                acc[1][nt] = wmma_f16(A2[1].h, B2.h, acc[1][nt]);
            }
        }
        __syncthreads();
    }

    // Epilogue: y = acc + b2, scatter gate*y into out.
#pragma unroll
    for (int ms = 0; ms < 2; ++ms)
#pragma unroll
        for (int r = 0; r < 8; ++r) {
            int m = ms * 16 + r + 8 * hlf;
            int tok = stok[m];
            if (tok < 0) continue;
            float g = sgate[m];
#pragma unroll
            for (int nt = 0; nt < 8; ++nt) {
                int ncol = nbase + nt * 16 + ln;
                float y = acc[ms][nt][r] + b2[e * ND + ncol];
                atomicAdd(&out[(size_t)tok * ND + ncol], g * y);
            }
        }
}

// ---------------------------------------------------------------------------
// Kernel 3: where(out == 0) -> x
// ---------------------------------------------------------------------------
__global__ void moe_finalize_kernel(const float* __restrict__ x,
                                    float* __restrict__ out, int n)
{
    int i = blockIdx.x * blockDim.x + threadIdx.x;
    if (i < n) {
        float v = out[i];
        if (v == 0.f) out[i] = x[i];
    }
}

extern "C" void kernel_launch(void* const* d_in, const int* in_sizes, int n_in,
                              void* d_out, int out_size, void* d_ws, size_t ws_size,
                              hipStream_t stream) {
    const float* x  = (const float*)d_in[0];  // [T, D]
    const float* rw = (const float*)d_in[1];  // [D, E]
    const float* w1 = (const float*)d_in[2];  // [E, D, F]
    const float* b1 = (const float*)d_in[3];  // [E, F]
    const float* w2 = (const float*)d_in[4];  // [E, F, D]
    const float* b2 = (const float*)d_in[5];  // [E, D]

    float* out      = (float*)d_out;               // [T, D]
    float* gatesOut = out + (size_t)NT * ND;       // [T, E]
    float* top1Out  = gatesOut + (size_t)NT * NE;  // [T] (as float)

    char*  ws       = (char*)d_ws;
    int*   cnt      = (int*)ws;                               // E
    int*   pairTok  = (int*)(ws + 256);                       // E*NT
    float* pairGate = (float*)(ws + 256 + (size_t)NE * NT * 4);

    // allow ~90 KB dynamic LDS (gfx1250: up to 320 KB per workgroup)
    hipFuncSetAttribute((const void*)moe_ffn_kernel,
                        hipFuncAttributeMaxDynamicSharedMemorySize, SMEM_BYTES);

    hipMemsetAsync(cnt, 0, NE * sizeof(int), stream);
    hipMemsetAsync(out, 0, (size_t)NT * ND * sizeof(float), stream);

    moe_router_kernel<<<NT / 8, 256, 0, stream>>>(
        x, rw, gatesOut, top1Out, cnt, pairTok, pairGate);

    moe_ffn_kernel<<<NE * TILES, 256, SMEM_BYTES, stream>>>(
        x, w1, b1, w2, b2, cnt, pairTok, pairGate, out);

    moe_finalize_kernel<<<(NT * ND + 255) / 256, 256, 0, stream>>>(
        x, out, NT * ND);
}
